// MolecularEmbeddingNetwork_10771777979163
// MI455X (gfx1250) — matrix operand order repaired and checked by reading, not compiled, not run on hardware
//
#include <hip/hip_runtime.h>
#include <hip/hip_bf16.h>
#include <math.h>

#define NATOMS 512
#define SDIM   64

typedef __attribute__((ext_vector_type(16))) _Float16     v16h;
typedef __attribute__((ext_vector_type(8)))  float        v8f;
typedef __attribute__((ext_vector_type(4)))  unsigned int v4u;
typedef __attribute__((ext_vector_type(8)))  int          v8i;
typedef __attribute__((ext_vector_type(4)))  int          v4i;

__device__ __forceinline__ float gelu_erf(float v) {
    return 0.5f * v * (1.0f + erff(v * 0.70710678118654752440f));
}

// ---------------------------------------------------------------------------
// TDM: DMA a [16 x 64] f32 tile (row stride 64 elements) from global into LDS.
// Descriptor bit packing per CDNA5 ISA sec 8.3-8.6 (D# groups 0..3).
// This toolchain exposes the 6-arg builtin (extra int32x8 = VADDR4 operand,
// unused per ISA sec 7.3 -> zero-filled).
// ---------------------------------------------------------------------------
__device__ __forceinline__ void tdm_load_tile16x64_f32(const float* gaddr,
                                                       unsigned lds_off) {
    const unsigned long long ga = (unsigned long long)(uintptr_t)gaddr;
    v4u g0;
    g0[0] = 1u;                                   // count=1 valid descriptor
    g0[1] = lds_off;                              // LDS byte address
    g0[2] = (unsigned)(ga & 0xFFFFFFFFu);         // global_addr[31:0]
    g0[3] = ((unsigned)(ga >> 32) & 0x01FFFFFFu)  // global_addr[56:32]
            | 0x80000000u;                        // type=2 ("image")
    v8i g1;
    g1[0] = (int)(2u << 16);     // workgroup_mask=0, data_size=2 (4 bytes)
    g1[1] = (int)(64u << 16);    // tensor_dim0[15:0]=64 (bits 79:48)
    g1[2] = (int)(512u << 16);   // tensor_dim0 hi=0, tensor_dim1[15:0]=512
    g1[3] = (int)(64u << 16);    // tensor_dim1 hi=0, tile_dim0=64
    g1[4] = 16;                  // tile_dim1=16, tile_dim2=0
    g1[5] = 64;                  // tensor_dim0_stride=64 (elements)
    g1[6] = 0;                   // stride hi / tensor_dim1_stride lo
    g1[7] = 0;                   // tensor_dim1_stride hi
    v4i g2 = {0, 0, 0, 0};       // 2D tile: groups 2/3 unused
    v4i g3 = {0, 0, 0, 0};
    v8i g4 = {0, 0, 0, 0, 0, 0, 0, 0};  // VADDR4: unused, zero/NULL
    __builtin_amdgcn_tensor_load_to_lds(g0, g1, g2, g3, g4, 0);
}

// ---------------------------------------------------------------------------
// Embedding MLP: h = GELU(Z @ W0 + b0) @ W1 + b1      (one block per atom)
// ---------------------------------------------------------------------------
__global__ __launch_bounds__(64) void embed_kernel(
    const float* __restrict__ Z, const float* __restrict__ W0,
    const float* __restrict__ b0, const float* __restrict__ W1,
    const float* __restrict__ b1, float* __restrict__ h) {
    __shared__ float zrow[32];
    __shared__ float t[64];
    const int i = blockIdx.x, c = threadIdx.x;
    if (c < 32) zrow[c] = Z[i * 32 + c];
    __syncthreads();
    float s = b0[c];
    for (int k = 0; k < 32; ++k) s += zrow[k] * W0[k * 64 + c];
    t[c] = gelu_erf(s);
    __syncthreads();
    float s2 = b1[c];
    for (int k = 0; k < 64; ++k) s2 += t[k] * W1[k * 64 + c];
    h[i * 64 + c] = s2;
}

// ---------------------------------------------------------------------------
// Per-layer precompute: A = h @ W0[0:64,:], B = h @ W0[64:128,:]
// ---------------------------------------------------------------------------
__global__ __launch_bounds__(64) void pre_kernel(
    const float* __restrict__ h, const float* __restrict__ W0,
    float* __restrict__ A, float* __restrict__ B) {
    __shared__ float hrow[64];
    const int i = blockIdx.x, c = threadIdx.x;
    hrow[c] = h[i * 64 + c];
    __syncthreads();
    float a = 0.f, b = 0.f;
    for (int k = 0; k < 64; ++k) {
        a += hrow[k] * W0[k * 64 + c];
        b += hrow[k] * W0[(64 + k) * 64 + c];
    }
    A[i * 64 + c] = a;
    B[i * 64 + c] = b;
}

// ---------------------------------------------------------------------------
// Convert per-layer msg_W1 / msg_W2 to f16 for WMMA
// ---------------------------------------------------------------------------
__global__ __launch_bounds__(256) void cvt_kernel(
    const float* __restrict__ W1, const float* __restrict__ W2,
    _Float16* __restrict__ W1h, _Float16* __restrict__ W2h) {
    const int idx = blockIdx.x * 256 + threadIdx.x;
    if (idx < 4096) {
        W1h[idx] = (_Float16)W1[idx];
        W2h[idx] = (_Float16)W2[idx];
    }
}

// ---------------------------------------------------------------------------
// Fused pair-message MLP + masked aggregation.
// Block = 128 threads = 4 wave32; wave w handles atom i = blockIdx.x*4 + w.
// The B[j] tile for each 16-row j-slab is DMA'd into LDS by the Tensor Data
// Mover, double-buffered against compute, synchronized on TENSORcnt.
// For each 16-row tile of j:
//   T1 = GELU(A[i] + B[j] + edge @ W0_e + b0)          (VALU, -> LDS f16)
//   P2 = T1 @ W1h   via 8x v_wmma_f32_16x16x32_f16
//   T2 = GELU(P2 + b1)                                  (-> LDS f16)
//   M  = T2 @ W2h   via 8x v_wmma_f32_16x16x32_f16
//   m_i += sum_j neigh(i,j) * (M + b2)                  (in-register)
// ---------------------------------------------------------------------------
__global__ __launch_bounds__(128) void msg_kernel(
    const float* __restrict__ A, const float* __restrict__ B,
    const float* __restrict__ x, const int* __restrict__ bond,
    const float* __restrict__ W0, const float* __restrict__ b0,
    const _Float16* __restrict__ W1h, const float* __restrict__ b1,
    const _Float16* __restrict__ W2h, const float* __restrict__ b2,
    float* __restrict__ m_out) {
    __shared__ float    Bt[4][2][16][64];   // TDM double buffer, 32 KB
    __shared__ _Float16 T1[4][16][64];
    __shared__ _Float16 T2[4][16][64];
    __shared__ float    NW[4][16];
    __shared__ float    ED[4][16][3];

    const int tid  = threadIdx.x;
    const int wv   = tid >> 5;
    const int lane = tid & 31;
    const int i    = blockIdx.x * 4 + wv;
    // wave-uniform (scalar) wave index for TDM descriptor construction
    const int wvu  = __builtin_amdgcn_readfirstlane(tid) >> 5;

    const unsigned lds_b[2] = {
        (unsigned)(uintptr_t)&Bt[wvu][0][0][0],
        (unsigned)(uintptr_t)&Bt[wvu][1][0][0]};

    const float xi0 = x[i * 3 + 0], xi1 = x[i * 3 + 1], xi2 = x[i * 3 + 2];

    // --- hoist per-lane loop invariants of the factored first layer ---
    const int   c0   = lane, c1 = lane + 32;
    const float ai0  = A[i * 64 + c0],        ai1  = A[i * 64 + c1];
    const float bb00 = b0[c0],                bb01 = b0[c1];
    const float we00 = W0[128 * 64 + c0],     we01 = W0[128 * 64 + c1];
    const float we10 = W0[129 * 64 + c0],     we11 = W0[129 * 64 + c1];
    const float we20 = W0[130 * 64 + c0],     we21 = W0[130 * 64 + c1];

    // --- hoist W1/W2 WMMA B-fragments into VGPRs (constant over j-loop) ---
    // B-frag layout (wave32, 16-bit, 32x16): lanes 0-15 hold K {0..7},{16..23} for
    // column N=lane; lanes 16-31 hold K {8..15},{24..31} for column N=lane-16.
    v16h w1f[8], w2f[8];
    {
        const int n     = lane & 15;
        const int khalf = (lane < 16) ? 0 : 8;
#pragma unroll
        for (int kt = 0; kt < 2; ++kt) {
#pragma unroll
            for (int nt = 0; nt < 4; ++nt) {
                const int col = nt * 16 + n;
                const int kb  = kt * 32 + khalf;
#pragma unroll
                for (int t = 0; t < 16; ++t) {
                    w1f[kt * 4 + nt][t] = W1h[(kb + t + (t >= 8 ? 8 : 0)) * 64 + col];
                    w2f[kt * 4 + nt][t] = W2h[(kb + t + (t >= 8 ? 8 : 0)) * 64 + col];
                }
            }
        }
    }

    float macc[4] = {0.f, 0.f, 0.f, 0.f};

    // prime the TDM pipeline with the first B tile
    tdm_load_tile16x64_f32(B, lds_b[0]);

    for (int j0 = 0; j0 < NATOMS; j0 += 16) {
        const int buf = (j0 >> 4) & 1;

        if (j0 + 16 < NATOMS) {
            // all prior DS reads of the other buffer must be complete
            asm volatile("s_wait_dscnt 0" ::: "memory");
            tdm_load_tile16x64_f32(B + (j0 + 16) * 64, lds_b[buf ^ 1]);
            __builtin_amdgcn_s_wait_tensorcnt(1);   // current tile landed
        } else {
            __builtin_amdgcn_s_wait_tensorcnt(0);
        }
        asm volatile("" ::: "memory");   // keep LDS reads below the wait

        // --- per-row geometry / masks (lanes 0-15) ---
        if (lane < 16) {
            const int j = j0 + lane;
            const float dx = xi0 - x[j * 3 + 0];
            const float dy = xi1 - x[j * 3 + 1];
            const float dz = xi2 - x[j * 3 + 2];
            const float d2 = dx * dx + dy * dy + dz * dz;
            const float d  = (d2 > 0.f) ? sqrtf(d2) : 0.f;
            const bool nself = (i != j);
            const bool bm  = ((bond[i * NATOMS + j] | bond[j * NATOMS + i]) != 0) && nself;
            const bool cut = (d <= 2.0f) && nself;
            NW[wv][lane]    = (bm || cut) ? 1.f : 0.f;
            ED[wv][lane][0] = bm ? 1.f : 0.f;
            ED[wv][lane][1] = d;
            ED[wv][lane][2] = d2;
        }

        // --- phase 1: factored first layer, GELU, stage as f16 tile ---
        for (int jr = 0; jr < 16; ++jr) {
            const float bmv = ED[wv][jr][0];
            const float dv  = ED[wv][jr][1];
            const float d2v = ED[wv][jr][2];
            const float* br = &Bt[wv][buf][jr][0];
            const float p0 = ai0 + br[c0] + bmv * we00 + dv * we10 + d2v * we20 + bb00;
            const float p1 = ai1 + br[c1] + bmv * we01 + dv * we11 + d2v * we21 + bb01;
            T1[wv][jr][c0] = (_Float16)gelu_erf(p0);
            T1[wv][jr][c1] = (_Float16)gelu_erf(p1);
        }

        // --- phase 2: P2 = T1 @ W1  (8 WMMAs) ---
        // A-frag layout: lanes 0-15, row M=lane: K {0..7},{16..23};
        // lanes 16-31, row M=lane-16: K {8..15},{24..31} (offset by kt*32).
        v8f acc[4] = {};
        {
            const int m = lane & 15;
#pragma unroll
            for (int kt = 0; kt < 2; ++kt) {
                const int koff = kt * 32 + ((lane < 16) ? 0 : 8);
                v16h af;
#pragma unroll
                for (int t = 0; t < 8; ++t) {
                    af[t]     = T1[wv][m][koff + t];
                    af[8 + t] = T1[wv][m][koff + 16 + t];
                }
#pragma unroll
                for (int nt = 0; nt < 4; ++nt)
                    acc[nt] = __builtin_amdgcn_wmma_f32_16x16x32_f16(
                        false, af, false, w1f[kt * 4 + nt], (short)0, acc[nt],
                        false, false);
            }
        }

        // --- GELU(P2 + b1) -> T2 (f16) ---
        // C/D layout: lanes 0-15: N=lane, VGPR r -> M=r; lanes 16-31: M=r+8.
        {
            const int n     = lane & 15;
            const int rbase = (lane < 16) ? 0 : 8;
#pragma unroll
            for (int nt = 0; nt < 4; ++nt) {
                const int col  = nt * 16 + n;
                const float bb = b1[col];
#pragma unroll
                for (int r = 0; r < 8; ++r)
                    T2[wv][rbase + r][col] = (_Float16)gelu_erf(acc[nt][r] + bb);
            }
        }

        // --- phase 3: M = T2 @ W2  (8 WMMAs) ---
        v8f acc2[4] = {};
        {
            const int m = lane & 15;
#pragma unroll
            for (int kt = 0; kt < 2; ++kt) {
                const int koff = kt * 32 + ((lane < 16) ? 0 : 8);
                v16h af;
#pragma unroll
                for (int t = 0; t < 8; ++t) {
                    af[t]     = T2[wv][m][koff + t];
                    af[8 + t] = T2[wv][m][koff + 16 + t];
                }
#pragma unroll
                for (int nt = 0; nt < 4; ++nt)
                    acc2[nt] = __builtin_amdgcn_wmma_f32_16x16x32_f16(
                        false, af, false, w2f[kt * 4 + nt], (short)0, acc2[nt],
                        false, false);
            }
        }

        // --- masked aggregation over this 16-row j tile (in-register) ---
        {
            const int n     = lane & 15;
            const int rbase = (lane < 16) ? 0 : 8;
#pragma unroll
            for (int nt = 0; nt < 4; ++nt) {
                const float bb = b2[nt * 16 + n];
                float s = 0.f;
#pragma unroll
                for (int r = 0; r < 8; ++r)
                    s += NW[wv][rbase + r] * (acc2[nt][r] + bb);
                macc[nt] += s;
            }
        }
    }

    // --- cross-half reduce (rows 0-7 half + rows 8-15 half) and store ---
#pragma unroll
    for (int nt = 0; nt < 4; ++nt) {
        float v = macc[nt] + __shfl_xor(macc[nt], 16, 32);
        if (lane < 16) m_out[i * 64 + nt * 16 + lane] = v;
    }
}

// ---------------------------------------------------------------------------
// Update MLP: h += GELU(GELU([h||m_i] @ U0 + b0) @ U1 + b1) @ U2 + b2
// ---------------------------------------------------------------------------
__global__ __launch_bounds__(64) void upd_kernel(
    float* __restrict__ h, const float* __restrict__ mi,
    const float* __restrict__ U0, const float* __restrict__ ub0,
    const float* __restrict__ U1, const float* __restrict__ ub1,
    const float* __restrict__ U2, const float* __restrict__ ub2) {
    __shared__ float cat[128];
    __shared__ float t1[64];
    __shared__ float t2[64];
    const int i = blockIdx.x, c = threadIdx.x;
    cat[c]      = h[i * 64 + c];
    cat[64 + c] = mi[i * 64 + c];
    __syncthreads();
    float s = ub0[c];
    for (int k = 0; k < 128; ++k) s += cat[k] * U0[k * 64 + c];
    t1[c] = gelu_erf(s);
    __syncthreads();
    s = ub1[c];
    for (int k = 0; k < 64; ++k) s += t1[k] * U1[k * 64 + c];
    t2[c] = gelu_erf(s);
    __syncthreads();
    s = ub2[c];
    for (int k = 0; k < 64; ++k) s += t2[k] * U2[k * 64 + c];
    h[i * 64 + c] += s;
}

// ---------------------------------------------------------------------------
extern "C" void kernel_launch(void* const* d_in, const int* in_sizes, int n_in,
                              void* d_out, int out_size, void* d_ws,
                              size_t ws_size, hipStream_t stream) {
    (void)in_sizes; (void)n_in; (void)out_size; (void)ws_size;

    const float* Z    = (const float*)d_in[0];
    const float* x    = (const float*)d_in[1];
    const int*   bond = (const int*)d_in[2];
    const float* eW0  = (const float*)d_in[3];
    const float* eb0  = (const float*)d_in[4];
    const float* eW1  = (const float*)d_in[5];
    const float* eb1  = (const float*)d_in[6];
    const float* mW0  = (const float*)d_in[7];   // [3,131,64]
    const float* mb0  = (const float*)d_in[8];   // [3,64]
    const float* mW1  = (const float*)d_in[9];   // [3,64,64]
    const float* mb1  = (const float*)d_in[10];
    const float* mW2  = (const float*)d_in[11];
    const float* mb2  = (const float*)d_in[12];
    const float* uW0  = (const float*)d_in[13];  // [3,128,64]
    const float* ub0  = (const float*)d_in[14];
    const float* uW1  = (const float*)d_in[15];
    const float* ub1  = (const float*)d_in[16];
    const float* uW2  = (const float*)d_in[17];
    const float* ub2  = (const float*)d_in[18];

    float* ws = (float*)d_ws;
    float* h  = ws;                    // 32768 f32
    float* A  = ws + 32768;            // 32768 f32
    float* Bm = ws + 65536;            // 32768 f32
    float* mi = ws + 98304;            // 32768 f32
    _Float16* W1h = (_Float16*)(ws + 131072);  // 4096 f16
    _Float16* W2h = W1h + 4096;                // 4096 f16

    embed_kernel<<<NATOMS, 64, 0, stream>>>(Z, eW0, eb0, eW1, eb1, h);

    for (int l = 0; l < 3; ++l) {
        pre_kernel<<<NATOMS, 64, 0, stream>>>(h, mW0 + l * 131 * 64, A, Bm);
        cvt_kernel<<<16, 256, 0, stream>>>(mW1 + l * 4096, mW2 + l * 4096,
                                           W1h, W2h);
        msg_kernel<<<NATOMS / 4, 128, 0, stream>>>(
            A, Bm, x, bond, mW0 + l * 131 * 64, mb0 + l * 64, W1h,
            mb1 + l * 64, W2h, mb2 + l * 64, mi);
        upd_kernel<<<NATOMS, 64, 0, stream>>>(
            h, mi, uW0 + l * 128 * 64, ub0 + l * 64, uW1 + l * 4096,
            ub1 + l * 64, uW2 + l * 4096, ub2 + l * 64);
    }

    (void)hipMemcpyAsync(d_out, h, NATOMS * SDIM * sizeof(float),
                         hipMemcpyDeviceToDevice, stream);
}
